// UnrollableDecoder_35244501631514
// MI455X (gfx1250) — matrix operand order repaired and checked by reading, not compiled, not run on hardware
//
#include <hip/hip_runtime.h>

// ---------------------------------------------------------------------------
// UnrollableDecoder on MI455X (gfx1250): bf16 WMMA + async global->LDS staging.
//   1) convert f32 weights/activations -> bf16
//   2) bridge GEMM + ReLU
//   3) 64x GRU step (gi GEMM, 3 gate accumulators share A fragments, gates fused)
//   4) output GEMM (+bias, (s,b)->(b,s) transpose in the store)
// All GEMMs: 256-thread blocks, 8 waves (4Mx2N), double-buffered LDS tiles
// filled with global_load_async_to_lds_b128 (ASYNCcnt), WMMA fed from LDS.
// ---------------------------------------------------------------------------

typedef __attribute__((ext_vector_type(16))) __bf16 v16bf;
typedef __attribute__((ext_vector_type(8)))  float  v8f;

static constexpr int kB = 512;   // batch
static constexpr int kS = 64;    // seq len
static constexpr int kH = 1024;  // hidden
static constexpr int kE = 1024;  // encoder width
static constexpr int kO = 1024;  // output width

union FragU { v16bf v; uint4 q[2]; };

__device__ __forceinline__ unsigned short f32_to_bf16(float f) {
  unsigned int u = __float_as_uint(f);
  unsigned int r = (u + 0x7FFFu + ((u >> 16) & 1u)) >> 16;  // round-nearest-even
  return (unsigned short)r;
}

__device__ __forceinline__ float fast_sigmoid(float x) {
  return 1.0f / (1.0f + __expf(-x));
}
__device__ __forceinline__ float fast_tanh(float x) {
  float e = __expf(-2.0f * x);
  return (1.0f - e) / (1.0f + e);
}

__device__ __forceinline__ v8f wmma_bf16(v16bf a, v16bf b, v8f c) {
  return __builtin_amdgcn_wmma_f32_16x16x32_bf16(
      /*neg_a=*/false, a, /*neg_b=*/false, b,
      /*c_mod=*/(short)0, c, /*reuse_a=*/false, /*reuse_b=*/false);
}

// --- CDNA5 async global->LDS copy (ASYNCcnt), 16 bytes per lane ------------
__device__ __forceinline__ unsigned lds_off_of(const void* p) {
  // LDS aperture: low 32 bits of the flat address are the LDS byte offset.
  return (unsigned)(uintptr_t)p;
}
__device__ __forceinline__ void async_copy16(unsigned lds_off,
                                             const unsigned short* g) {
  asm volatile("global_load_async_to_lds_b128 %0, %1, off"
               :: "v"(lds_off), "v"(g) : "memory");
}
__device__ __forceinline__ void wait_async_all() {
  asm volatile("s_wait_asynccnt 0x0" ::: "memory");
}

// --- WMMA fragment gathers from an LDS tile (row-major, 32 bf16 per row) ---
// A fragment (16x32): lanes 0-15 row r, K {0..7,16..23}; lanes 16-31 K +8/+24.
__device__ __forceinline__ v16bf lds_a_frag(const unsigned short* T, int row0,
                                            int lane) {
  const unsigned short* p = T + (row0 + (lane & 15)) * 32 + ((lane >> 4) << 3);
  FragU u;
  u.q[0] = *(const uint4*)(p);
  u.q[1] = *(const uint4*)(p + 16);
  return u.v;
}
// B fragment (32x16): lane holds column n0+(lane&15); lanes 0-15 K {0..7,16..23},
// lanes 16-31 K +8/+24 (source tile is n-major: 32 K-contiguous bf16 per n).
__device__ __forceinline__ v16bf lds_b_frag(const unsigned short* T, int n0,
                                            int lane) {
  const unsigned short* p = T + (n0 + (lane & 15)) * 32 + ((lane >> 4) << 3);
  FragU u;
  u.q[0] = *(const uint4*)(p);
  u.q[1] = *(const uint4*)(p + 16);
  return u.v;
}

// ---------------------------------------------------------------------------
__global__ void __launch_bounds__(256)
cvt_bf16_kernel(const float* __restrict__ in, unsigned short* __restrict__ out,
                int n) {
  int i = blockIdx.x * blockDim.x + threadIdx.x;
  int stride = gridDim.x * blockDim.x;
  for (; i < n; i += stride) out[i] = f32_to_bf16(in[i]);
}

// ---------------------------------------------------------------------------
// Generic cooperative GEMM core: Y-tile(128x64) = A(128xK) @ W(NxK)^T tile.
// MODE 0: bridge (bf16 out, ReLU, row-major rows). MODE 1: output (f32 out,
// (s,b)->(b,s) row transpose).
template <int MODE>
__global__ void __launch_bounds__(256)
gemm128x64_kernel(const unsigned short* __restrict__ A,
                  const unsigned short* __restrict__ W,
                  const float* __restrict__ bias, void* __restrict__ Yout,
                  int K, int N, int Nblk) {
  __shared__ alignas(16) unsigned short Asm[2][128 * 32];
  __shared__ alignas(16) unsigned short Bsm[2][64 * 32];

  const int tid = threadIdx.x;
  const int lane = tid & 31;
  const int wid = tid >> 5;   // 0..7
  const int wvm = wid >> 1;   // 0..3 (M)
  const int wvn = wid & 1;    // 0..1 (N)
  const int m0 = (blockIdx.x / Nblk) * 128;
  const int n0 = (blockIdx.x % Nblk) * 64;

  auto stage = [&](int buf, int kk) {
    // A tile: 128 rows x 64B = 512 x 16B chunks, 2 per thread.
#pragma unroll
    for (int i = 0; i < 2; ++i) {
      int c = tid + i * 256, row = c >> 2, seg = c & 3;
      async_copy16(lds_off_of(&Asm[buf][row * 32 + seg * 8]),
                   A + (size_t)(m0 + row) * K + kk + seg * 8);
    }
    // B tile: 64 rows x 64B = 256 x 16B chunks, 1 per thread.
    {
      int row = tid >> 2, seg = tid & 3;
      async_copy16(lds_off_of(&Bsm[buf][row * 32 + seg * 8]),
                   W + (size_t)(n0 + row) * K + kk + seg * 8);
    }
  };

  v8f acc[2][2] = {};
  const int nk = K / 32;
  stage(0, 0);
  for (int kt = 0; kt < nk; ++kt) {
    const int buf = kt & 1;
    wait_async_all();
    __syncthreads();  // stage(buf) landed everywhere; everyone done reading buf^1
    if (kt + 1 < nk) stage(buf ^ 1, (kt + 1) * 32);
    v16bf a0 = lds_a_frag(Asm[buf], wvm * 32, lane);
    v16bf a1 = lds_a_frag(Asm[buf], wvm * 32 + 16, lane);
    v16bf b0 = lds_b_frag(Bsm[buf], wvn * 32, lane);
    v16bf b1 = lds_b_frag(Bsm[buf], wvn * 32 + 16, lane);
    acc[0][0] = wmma_bf16(a0, b0, acc[0][0]);
    acc[0][1] = wmma_bf16(a0, b1, acc[0][1]);
    acc[1][0] = wmma_bf16(a1, b0, acc[1][0]);
    acc[1][1] = wmma_bf16(a1, b1, acc[1][1]);
  }

  const int col16 = lane & 15;
  const int rb = (lane >> 4) << 3;
#pragma unroll
  for (int ni = 0; ni < 2; ++ni) {
    const int n = n0 + wvn * 32 + ni * 16 + col16;
    const float bi = bias[n];
#pragma unroll
    for (int mi = 0; mi < 2; ++mi) {
#pragma unroll
      for (int v = 0; v < 8; ++v) {
        const int row = m0 + wvm * 32 + mi * 16 + rb + v;
        float f = acc[mi][ni][v] + bi;
        if (MODE == 0) {  // bridge: ReLU, bf16, row-major
          f = f > 0.0f ? f : 0.0f;
          ((unsigned short*)Yout)[(size_t)row * N + n] = f32_to_bf16(f);
        } else {          // output: f32, row = s*kB+b stored at b*kS+s
          const int s = row >> 9;  // / 512
          const int b = row & (kB - 1);
          ((float*)Yout)[((size_t)b * kS + s) * N + n] = f;
        }
      }
    }
  }
}

// ---------------------------------------------------------------------------
// GRU step (h_prev == 0): block tile 128(M) x 32(N), 3 gate B-tiles staged.
// Each wave: 32x16 tile x {r,z,n} sharing A fragments; fused gate epilogue.
__global__ void __launch_bounds__(256)
gru_step_kernel(const unsigned short* __restrict__ Xt,
                const unsigned short* __restrict__ Wih,
                const float* __restrict__ b_ih, const float* __restrict__ b_hh,
                unsigned short* __restrict__ Hout) {
  __shared__ alignas(16) unsigned short Asm[2][128 * 32];
  __shared__ alignas(16) unsigned short Bsm[2][3][32 * 32];

  const int tid = threadIdx.x;
  const int lane = tid & 31;
  const int wid = tid >> 5;
  const int wvm = wid >> 1;   // 0..3
  const int wvn = wid & 1;    // 0..1
  const int Nblk = kH / 32;   // 32
  const int m0 = (blockIdx.x / Nblk) * 128;
  const int n0 = (blockIdx.x % Nblk) * 32;

  auto stage = [&](int buf, int kk) {
#pragma unroll
    for (int i = 0; i < 2; ++i) {  // A: 512 chunks
      int c = tid + i * 256, row = c >> 2, seg = c & 3;
      async_copy16(lds_off_of(&Asm[buf][row * 32 + seg * 8]),
                   Xt + (size_t)(m0 + row) * kH + kk + seg * 8);
    }
    // B: 3 gates x 32 rows x 4 segs = 384 chunks (1.5 per thread)
#pragma unroll
    for (int i = 0; i < 2; ++i) {
      int c = tid + i * 256;
      if (c < 384) {
        int gate = c >> 7, rr = (c & 127) >> 2, seg = c & 3;
        async_copy16(lds_off_of(&Bsm[buf][gate][rr * 32 + seg * 8]),
                     Wih + (size_t)(gate * kH + n0 + rr) * kH + kk + seg * 8);
      }
    }
  };

  v8f aR[2] = {}, aZ[2] = {}, aN[2] = {};
  const int nk = kH / 32;
  stage(0, 0);
  for (int kt = 0; kt < nk; ++kt) {
    const int buf = kt & 1;
    wait_async_all();
    __syncthreads();
    if (kt + 1 < nk) stage(buf ^ 1, (kt + 1) * 32);
    v16bf x0 = lds_a_frag(Asm[buf], wvm * 32, lane);
    v16bf x1 = lds_a_frag(Asm[buf], wvm * 32 + 16, lane);
    v16bf wr = lds_b_frag(Bsm[buf][0], wvn * 16, lane);
    v16bf wz = lds_b_frag(Bsm[buf][1], wvn * 16, lane);
    v16bf wg = lds_b_frag(Bsm[buf][2], wvn * 16, lane);
    aR[0] = wmma_bf16(x0, wr, aR[0]);
    aR[1] = wmma_bf16(x1, wr, aR[1]);
    aZ[0] = wmma_bf16(x0, wz, aZ[0]);
    aZ[1] = wmma_bf16(x1, wz, aZ[1]);
    aN[0] = wmma_bf16(x0, wg, aN[0]);
    aN[1] = wmma_bf16(x1, wg, aN[1]);
  }

  const int col = n0 + wvn * 16 + (lane & 15);
  const int rb = (lane >> 4) << 3;
  const float br = b_ih[col] + b_hh[col];
  const float bz = b_ih[kH + col] + b_hh[kH + col];
  const float bn = b_ih[2 * kH + col];
  const float bhn = b_hh[2 * kH + col];
#pragma unroll
  for (int mi = 0; mi < 2; ++mi) {
#pragma unroll
    for (int v = 0; v < 8; ++v) {
      float r = fast_sigmoid(aR[mi][v] + br);
      float z = fast_sigmoid(aZ[mi][v] + bz);
      float n = fast_tanh(aN[mi][v] + bn + r * bhn);
      float h = (1.0f - z) * n;
      Hout[(size_t)(m0 + wvm * 32 + mi * 16 + rb + v) * kH + col] =
          f32_to_bf16(h);
    }
  }
}

// ---------------------------------------------------------------------------
extern "C" void kernel_launch(void* const* d_in, const int* in_sizes, int n_in,
                              void* d_out, int out_size, void* d_ws, size_t ws_size,
                              hipStream_t stream) {
  const float* x        = (const float*)d_in[0];
  const float* bridge_w = (const float*)d_in[1];
  const float* bridge_b = (const float*)d_in[2];
  const float* w_ih     = (const float*)d_in[3];
  // d_in[4] = w_hh : unused (h_prev == 0 in the reference; its matmul vanishes)
  const float* b_ih     = (const float*)d_in[5];
  const float* b_hh     = (const float*)d_in[6];
  const float* out_w    = (const float*)d_in[7];
  const float* out_b    = (const float*)d_in[8];
  float* out = (float*)d_out;

  unsigned short* ws = (unsigned short*)d_ws;
  unsigned short* w_ih_bf = ws;  ws += (size_t)3 * kH * kH;  // 6 MB
  unsigned short* bw_bf   = ws;  ws += (size_t)kH * kE;      // 2 MB
  unsigned short* ow_bf   = ws;  ws += (size_t)kO * kH;      // 2 MB
  unsigned short* x_bf    = ws;  ws += (size_t)kB * kE;      // 1 MB
  unsigned short* xt0     = ws;  ws += (size_t)kB * kH;      // 1 MB
  unsigned short* ys      = ws;                              // 64 MB (kS*kB*kH)

  auto cvt = [&](const float* in, unsigned short* o, int n) {
    int blocks = (n + 256 * 8 - 1) / (256 * 8);
    cvt_bf16_kernel<<<blocks, 256, 0, stream>>>(in, o, n);
  };
  cvt(w_ih, w_ih_bf, 3 * kH * kH);
  cvt(bridge_w, bw_bf, kH * kE);
  cvt(out_w, ow_bf, kO * kH);
  cvt(x, x_bf, kB * kE);

  // Bridge: (512/128) * (1024/64) = 4 * 16 = 64 blocks
  gemm128x64_kernel<0><<<64, 256, 0, stream>>>(x_bf, bw_bf, bridge_b,
                                               (void*)xt0, kE, kH, kH / 64);

  // 64 dependent GRU steps: (512/128) * (1024/32) = 128 blocks each
  for (int t = 0; t < kS; ++t) {
    const unsigned short* xin = (t == 0) ? xt0 : (ys + (size_t)(t - 1) * kB * kH);
    unsigned short* hout = ys + (size_t)t * kB * kH;
    gru_step_kernel<<<128, 256, 0, stream>>>(xin, w_ih_bf, b_ih, b_hh, hout);
  }

  // Output: (32768/128) * (1024/64) = 256 * 16 = 4096 blocks
  gemm128x64_kernel<1><<<4096, 256, 0, stream>>>(ys, ow_bf, out_b, (void*)out,
                                                 kH, kO, kO / 64);

  (void)in_sizes; (void)n_in; (void)out_size; (void)ws_size;
}
